// AdaptiveHierarchicalPrototypes_6837587935767
// MI455X (gfx1250) — matrix operand order repaired and checked by reading, not compiled, not run on hardware
//
#include <hip/hip_runtime.h>
#include <hip/hip_bf16.h>

// ---------------------------------------------------------------------------
// Problem constants (from reference)
// ---------------------------------------------------------------------------
#define N_SUP   16384
#define DIM     1024
#define LVL     4
#define NCLS    64

// Main fused-GEMM tiling
#define BM       32                    // rows per workgroup
#define PITCH_BF 2064                  // bytes: 1024 bf16 + 16B pad (conflict-free frag loads)
#define PITCH_F  4112                  // bytes: 1024 f32 + 16B pad
#define OFF_HHF  (BM * PITCH_BF)       // 66048
#define OFF_LAB  (OFF_HHF + BM * PITCH_F)  // 197632 : 32 sorted labels
#define OFF_PRM  (OFF_LAB + 128)           // 197760 : 32 permuted row ids
#define OFF_RUN  (OFF_PRM + 128)           // 197888 : runStart[33] + nrun
#define SMEM_BYTES (OFF_RUN + 144)         // 198032  (CDNA5 WGP LDS = 320KB)

// ---------------------------------------------------------------------------
// Types for WMMA
// ---------------------------------------------------------------------------
typedef __attribute__((ext_vector_type(16))) __bf16       v16bf;
typedef __attribute__((ext_vector_type(8)))  float        v8f;
typedef __attribute__((ext_vector_type(4)))  unsigned int v4u;

union Frag {
    v4u   u4[2];
    v16bf bf;
};

__device__ __forceinline__ unsigned short f2bf(float f) {
    unsigned int u = __float_as_uint(f);
    u += 0x7FFFu + ((u >> 16) & 1u);   // round-to-nearest-even
    return (unsigned short)(u >> 16);
}

__device__ __forceinline__ void load_bquad(const unsigned* __restrict__ bp,
                                           Frag& b0, Frag& b1, Frag& b2, Frag& b3) {
    b0.u4[0] = *(const v4u*)bp;         b0.u4[1] = *(const v4u*)(bp + 4);
    b1.u4[0] = *(const v4u*)(bp + 256); b1.u4[1] = *(const v4u*)(bp + 260);
    b2.u4[0] = *(const v4u*)(bp + 512); b2.u4[1] = *(const v4u*)(bp + 516);
    b3.u4[0] = *(const v4u*)(bp + 768); b3.u4[1] = *(const v4u*)(bp + 772);
}

// ---------------------------------------------------------------------------
// K0: column sums of support_features (partial, atomic) -> tc (unnormalized)
// ---------------------------------------------------------------------------
__global__ void __launch_bounds__(256) colsum_kernel(const float* __restrict__ X,
                                                     float* __restrict__ tc) {
    int col = blockIdx.x * 256 + threadIdx.x;        // 0..1023 (gridDim.x = 4)
    int r0  = blockIdx.y * 1024;                     // gridDim.y = 16
    const float* p = X + (size_t)r0 * DIM + col;
    float s = 0.f;
    for (int r = 0; r < 1024; ++r) s += p[(size_t)r * DIM];
    atomicAdd(&tc[col], s);
}

// ---------------------------------------------------------------------------
// K1: selector MLP: acts = sigmoid(relu(tc@sp_w1+b1)@sp_w2+b2); mask/slots/k
// ---------------------------------------------------------------------------
__global__ void __launch_bounds__(512) splevel_kernel(
    const float* __restrict__ tc, const float* __restrict__ w1,
    const float* __restrict__ b1, const float* __restrict__ w2,
    const float* __restrict__ b2, float* __restrict__ actsW,
    int* __restrict__ maskinfo, float* __restrict__ outActs) {
    __shared__ float stc[1024];
    __shared__ float sh[512];
    __shared__ float sacts[4];
    int t = threadIdx.x;
    stc[t]       = tc[t]       * (1.f / (float)N_SUP);
    stc[t + 512] = tc[t + 512] * (1.f / (float)N_SUP);
    __syncthreads();
    float acc = b1[t];
    for (int k = 0; k < 1024; ++k) acc = fmaf(stc[k], w1[k * 512 + t], acc);
    sh[t] = fmaxf(acc, 0.f);
    __syncthreads();
    if (t < 4) {
        float a = b2[t];
        for (int k = 0; k < 512; ++k) a = fmaf(sh[k], w2[k * 4 + t], a);
        a = 1.f / (1.f + __expf(-a));
        sacts[t]     = a;
        actsW[t]     = a;
        outActs[t]   = a;                         // acts
        outActs[4+t] = (a > 0.1f) ? a : 0.f;      // level_contributions
    }
    __syncthreads();
    if (t == 0) {
        int k = 0;
        for (int l = 0; l < LVL; ++l) {
            int m = (sacts[l] > 0.1f) ? 1 : 0;
            maskinfo[1 + l] = m ? k : -1;         // slot or -1
            k += m;
        }
        maskinfo[0] = k;
    }
}

// ---------------------------------------------------------------------------
// Counting sort of rows by class: hist -> scan -> scatter
// ---------------------------------------------------------------------------
__global__ void __launch_bounds__(256) hist_kernel(const int* __restrict__ lab,
                                                   int* __restrict__ icnt) {
    int r = blockIdx.x * 256 + threadIdx.x;          // 64 blocks
    atomicAdd(&icnt[lab[r]], 1);
}

__global__ void scan_kernel(const int* __restrict__ icnt, int* __restrict__ binOff,
                            float* __restrict__ cnts) {
    if (threadIdx.x == 0) {
        int acc = 0;
        for (int c = 0; c < NCLS; ++c) {
            binOff[c] = acc;
            acc += icnt[c];
            cnts[c] = (float)icnt[c];
        }
        binOff[NCLS] = acc;
    }
}

__global__ void __launch_bounds__(256) scatter_kernel(
    const int* __restrict__ lab, const int* __restrict__ binOff,
    int* __restrict__ cursor, int* __restrict__ perm, int* __restrict__ labS) {
    int r = blockIdx.x * 256 + threadIdx.x;          // 64 blocks
    int cls = lab[r];
    int pos = binOff[cls] + atomicAdd(&cursor[cls], 1);
    perm[pos] = r;
    labS[pos] = cls;
}

// ---------------------------------------------------------------------------
// K2: raw segment sum over sorted rows: one atomic per (class,col) per run
// ---------------------------------------------------------------------------
__global__ void __launch_bounds__(256) rawsum_kernel(
    const float* __restrict__ X, const int* __restrict__ perm,
    const int* __restrict__ labS, float* __restrict__ rawSum) {
    __shared__ int sperm[BM];
    __shared__ int sl[BM];
    __shared__ int runStart[BM + 1];
    __shared__ int nrun;
    int base = blockIdx.x * BM;                      // 512 blocks
    int t = threadIdx.x;
    if (t < BM) { sperm[t] = perm[base + t]; sl[t] = labS[base + t]; }
    __syncthreads();
    if (t == 0) {
        int nr = 0;
        runStart[0] = 0;
        for (int i = 1; i < BM; ++i)
            if (sl[i] != sl[i - 1]) runStart[++nr] = i;
        runStart[nr + 1] = BM;
        nrun = nr + 1;
    }
    __syncthreads();
    for (int run = 0; run < nrun; ++run) {
        int s0 = runStart[run], s1 = runStart[run + 1];
        int cls = sl[s0];
        for (int col = t; col < DIM; col += 256) {
            float a = 0.f;
            for (int r = s0; r < s1; ++r)
                a += X[(size_t)sperm[r] * DIM + col];
            atomicAdd(&rawSum[(size_t)cls * DIM + col], a);
        }
    }
}

// ---------------------------------------------------------------------------
// K3: prepack weights into WMMA B-fragment order (bf16, packed 2/dword).
// ---------------------------------------------------------------------------
__global__ void __launch_bounds__(256) prepack_kernel(
    const float* __restrict__ W, unsigned* __restrict__ P, int lvlStride) {
    int t    = blockIdx.x * 256 + threadIdx.x;   // 2,097,152 total
    int v    = t & 7;
    int lane = (t >> 3) & 31;
    int nt   = (t >> 8) & 63;
    int kt   = (t >> 14) & 31;
    int l    = t >> 19;
    int k    = kt * 32 + ((lane >> 4) << 4) + 2 * v;
    int n    = (nt << 4) + (lane & 15);
    const float* base = W + (size_t)l * lvlStride;
    unsigned lo = f2bf(base[(size_t)k * DIM + n]);
    unsigned hi = f2bf(base[(size_t)(k + 1) * DIM + n]);
    P[t] = lo | (hi << 16);
}

// K3b: fold lvl_emb * (last row of an_w1) + an_b1 into c1
__global__ void __launch_bounds__(256) c1_kernel(
    const float* __restrict__ an_w1, const float* __restrict__ an_b1,
    const float* __restrict__ lvl_emb, float* __restrict__ c1) {
    int t = blockIdx.x * 256 + threadIdx.x;      // 4096
    int l = t >> 10, n = t & 1023;
    c1[t] = an_b1[t] + lvl_emb[l] * an_w1[(size_t)l * (1025 * DIM) + 1024 * DIM + n];
}

// ---------------------------------------------------------------------------
// K4: main fused per-level adapter over label-sorted rows:
//   GEMM1 (WMMA bf16, quad N-tile, double-buffered B) -> layernorm+relu
//   -> GEMM2 (WMMA bf16) -> per-run LDS reduction -> few atomics.
//   grid = (512 row-blocks, 4 levels), block = 256 (8 waves)
// ---------------------------------------------------------------------------
__global__ void __launch_bounds__(256) anet_kernel(
    const float* __restrict__ X, const int* __restrict__ perm,
    const int* __restrict__ labS, const unsigned* __restrict__ W1p,
    const unsigned* __restrict__ W2p, const float* __restrict__ c1,
    const float* __restrict__ gamma, const float* __restrict__ beta,
    const float* __restrict__ b2, float* __restrict__ protoSum) {
    extern __shared__ char smem[];
    const int rb   = blockIdx.x;
    const int l    = blockIdx.y;
    const int tid  = threadIdx.x;
    const int wave = tid >> 5;
    const int lane = tid & 31;

    int* slab     = (int*)(smem + OFF_LAB);
    int* sperm    = (int*)(smem + OFF_PRM);
    int* runStart = (int*)(smem + OFF_RUN);
    int* nrunp    = (int*)(smem + OFF_RUN + (BM + 1) * 4);

    if (tid < BM) {
        slab[tid]  = labS[rb * BM + tid];
        sperm[tid] = perm[rb * BM + tid];
    }
    __syncthreads();

    if (tid == 0) {                        // run detection over sorted labels
        int nr = 0;
        runStart[0] = 0;
        for (int i = 1; i < BM; ++i)
            if (slab[i] != slab[i - 1]) runStart[++nr] = i;
        runStart[nr + 1] = BM;
        *nrunp = nr + 1;
    }

    // stage permuted X rows -> LDS bf16 (coalesced per-row global reads)
    for (int idx = tid; idx < BM * DIM; idx += 256) {
        int r = idx >> 10, cc = idx & 1023;
        float xv = X[(size_t)sperm[r] * DIM + cc];
        *(unsigned short*)(smem + r * PITCH_BF + cc * 2) = f2bf(xv);
    }
    __syncthreads();

    const int m_w   = wave & 1;            // M-tile (2 x 16 rows)
    const int ng    = wave >> 1;           // N-quad group 0..3
    const int arow  = m_w * 16 + (lane & 15);
    const int asel  = (lane >> 4) << 4;    // A frag K-subblock select
    const int ncol  = lane & 15;
    const int crow0 = m_w * 16 + ((lane >> 4) << 3);
    const char* abase = smem + arow * PITCH_BF + asel * 2;

    // ---------------- GEMM1: hh = Xtile @ W1 + c1 (quad-N, dbl-buf B) ----
    for (int j = 0; j < 4; ++j) {
        int nt0 = (((j << 2) + ng) << 2);  // quad base tile: 0,4,...,60
        v8f acc0 = {}, acc1 = {}, acc2 = {}, acc3 = {};
        const unsigned* bp = W1p + (((size_t)(l * 2048 + nt0)) << 8) + lane * 8;
        Frag B0, B1, B2, B3;
        load_bquad(bp, B0, B1, B2, B3);
        for (int kt = 0; kt < 31; ++kt) {
            bp += 64 * 256;
            Frag N0, N1, N2, N3;
            load_bquad(bp, N0, N1, N2, N3);          // prefetch next kt
            const char* ap = abase + kt * 64;
            Frag A; A.u4[0] = *(const v4u*)ap; A.u4[1] = *(const v4u*)(ap + 16);
            acc0 = __builtin_amdgcn_wmma_f32_16x16x32_bf16(
                false, A.bf, false, B0.bf, (short)0, acc0, false, false);
            acc1 = __builtin_amdgcn_wmma_f32_16x16x32_bf16(
                false, A.bf, false, B1.bf, (short)0, acc1, false, false);
            acc2 = __builtin_amdgcn_wmma_f32_16x16x32_bf16(
                false, A.bf, false, B2.bf, (short)0, acc2, false, false);
            acc3 = __builtin_amdgcn_wmma_f32_16x16x32_bf16(
                false, A.bf, false, B3.bf, (short)0, acc3, false, false);
            B0 = N0; B1 = N1; B2 = N2; B3 = N3;      // rotate (SSA, no copies)
        }
        {   // peeled last kt = 31
            const char* ap = abase + 31 * 64;
            Frag A; A.u4[0] = *(const v4u*)ap; A.u4[1] = *(const v4u*)(ap + 16);
            acc0 = __builtin_amdgcn_wmma_f32_16x16x32_bf16(
                false, A.bf, false, B0.bf, (short)0, acc0, false, false);
            acc1 = __builtin_amdgcn_wmma_f32_16x16x32_bf16(
                false, A.bf, false, B1.bf, (short)0, acc1, false, false);
            acc2 = __builtin_amdgcn_wmma_f32_16x16x32_bf16(
                false, A.bf, false, B2.bf, (short)0, acc2, false, false);
            acc3 = __builtin_amdgcn_wmma_f32_16x16x32_bf16(
                false, A.bf, false, B3.bf, (short)0, acc3, false, false);
        }
        int colg0 = (nt0 << 4) + ncol;
        float cb0 = c1[(l << 10) + colg0];
        float cb1 = c1[(l << 10) + colg0 + 16];
        float cb2 = c1[(l << 10) + colg0 + 32];
        float cb3 = c1[(l << 10) + colg0 + 48];
        char* hf0 = smem + OFF_HHF + colg0 * 4;
#pragma unroll
        for (int v = 0; v < 8; ++v) {
            char* rowp = hf0 + (crow0 + v) * PITCH_F;
            *(float*)(rowp)       = acc0[v] + cb0;
            *(float*)(rowp + 64)  = acc1[v] + cb1;
            *(float*)(rowp + 128) = acc2[v] + cb2;
            *(float*)(rowp + 192) = acc3[v] + cb3;
        }
    }
    __syncthreads();

    // ---------------- LayerNorm + ReLU -> bf16 (reuse X-tile LDS) --------
    for (int rr = 0; rr < 4; ++rr) {
        int r = (wave << 2) + rr;
        const float* hp = (const float*)(smem + OFF_HHF + r * PITCH_F);
        float s = 0.f, s2 = 0.f;
        for (int c = lane; c < DIM; c += 32) { float x = hp[c]; s += x; s2 += x * x; }
#pragma unroll
        for (int m = 16; m; m >>= 1) {
            s  += __shfl_xor(s,  m, 32);
            s2 += __shfl_xor(s2, m, 32);
        }
        float mean = s * (1.f / (float)DIM);
        float var  = fmaxf(s2 * (1.f / (float)DIM) - mean * mean, 0.f);
        float inv  = rsqrtf(var + 1e-5f);
        unsigned short* ob = (unsigned short*)(smem + r * PITCH_BF);
        for (int c = lane; c < DIM; c += 32) {
            float x = (hp[c] - mean) * inv * gamma[(l << 10) + c] + beta[(l << 10) + c];
            ob[c] = f2bf(fmaxf(x, 0.f));
        }
    }
    __syncthreads();

    // ---------------- GEMM2: t = hh @ W2 + b2 -> hhF (LDS) ---------------
    for (int j = 0; j < 4; ++j) {
        int nt0 = (((j << 2) + ng) << 2);
        v8f acc0 = {}, acc1 = {}, acc2 = {}, acc3 = {};
        const unsigned* bp = W2p + (((size_t)(l * 2048 + nt0)) << 8) + lane * 8;
        Frag B0, B1, B2, B3;
        load_bquad(bp, B0, B1, B2, B3);
        for (int kt = 0; kt < 31; ++kt) {
            bp += 64 * 256;
            Frag N0, N1, N2, N3;
            load_bquad(bp, N0, N1, N2, N3);
            const char* ap = abase + kt * 64;
            Frag A; A.u4[0] = *(const v4u*)ap; A.u4[1] = *(const v4u*)(ap + 16);
            acc0 = __builtin_amdgcn_wmma_f32_16x16x32_bf16(
                false, A.bf, false, B0.bf, (short)0, acc0, false, false);
            acc1 = __builtin_amdgcn_wmma_f32_16x16x32_bf16(
                false, A.bf, false, B1.bf, (short)0, acc1, false, false);
            acc2 = __builtin_amdgcn_wmma_f32_16x16x32_bf16(
                false, A.bf, false, B2.bf, (short)0, acc2, false, false);
            acc3 = __builtin_amdgcn_wmma_f32_16x16x32_bf16(
                false, A.bf, false, B3.bf, (short)0, acc3, false, false);
            B0 = N0; B1 = N1; B2 = N2; B3 = N3;
        }
        {   // peeled last kt = 31
            const char* ap = abase + 31 * 64;
            Frag A; A.u4[0] = *(const v4u*)ap; A.u4[1] = *(const v4u*)(ap + 16);
            acc0 = __builtin_amdgcn_wmma_f32_16x16x32_bf16(
                false, A.bf, false, B0.bf, (short)0, acc0, false, false);
            acc1 = __builtin_amdgcn_wmma_f32_16x16x32_bf16(
                false, A.bf, false, B1.bf, (short)0, acc1, false, false);
            acc2 = __builtin_amdgcn_wmma_f32_16x16x32_bf16(
                false, A.bf, false, B2.bf, (short)0, acc2, false, false);
            acc3 = __builtin_amdgcn_wmma_f32_16x16x32_bf16(
                false, A.bf, false, B3.bf, (short)0, acc3, false, false);
        }
        int colg0 = (nt0 << 4) + ncol;
        float bb0 = b2[(l << 10) + colg0];
        float bb1 = b2[(l << 10) + colg0 + 16];
        float bb2 = b2[(l << 10) + colg0 + 32];
        float bb3 = b2[(l << 10) + colg0 + 48];
        char* hf0 = smem + OFF_HHF + colg0 * 4;
#pragma unroll
        for (int v = 0; v < 8; ++v) {
            char* rowp = hf0 + (crow0 + v) * PITCH_F;
            *(float*)(rowp)       = acc0[v] + bb0;
            *(float*)(rowp + 64)  = acc1[v] + bb1;
            *(float*)(rowp + 128) = acc2[v] + bb2;
            *(float*)(rowp + 192) = acc3[v] + bb3;
        }
    }
    __syncthreads();

    // ---------------- per-run reduction + few atomics ---------------------
    int nrun = *nrunp;
    for (int run = 0; run < nrun; ++run) {
        int s0 = runStart[run], s1 = runStart[run + 1];
        int cls = slab[s0];
        float* dst = protoSum + ((((size_t)l << 6) + cls) << 10);
        for (int col = tid; col < DIM; col += 256) {
            float a = 0.f;
            for (int r = s0; r < s1; ++r)
                a += *(const float*)(smem + OFF_HHF + r * PITCH_F + col * 4);
            atomicAdd(&dst[col], a);
        }
    }
}

// ---------------------------------------------------------------------------
// K5: fusion MLP per class + final prototype select
// ---------------------------------------------------------------------------
__global__ void __launch_bounds__(256) fusion_kernel(
    const float* __restrict__ protoSum, const float* __restrict__ rawSum,
    const float* __restrict__ counts, const float* __restrict__ actsW,
    const int* __restrict__ maskinfo, const float* __restrict__ fw1,
    const float* __restrict__ fb1, const float* __restrict__ fw2,
    const float* __restrict__ fb2, float* __restrict__ outProto) {
    __shared__ float sp[LVL][DIM];
    __shared__ float sh[DIM];
    int c = blockIdx.x, t = threadIdx.x;
    float cnt  = fmaxf(counts[c], 1.f);
    float rinv = 1.f / cnt;
    int k = maskinfo[0];
    if (k == 0) {
        for (int j = t; j < DIM; j += 256)
            outProto[(size_t)c * DIM + j] = rawSum[(size_t)c * DIM + j] * rinv;
        return;
    }
    for (int l = 0; l < LVL; ++l)
        for (int j = t; j < DIM; j += 256)
            sp[l][j] = protoSum[(((l << 6) + c) << 10) + j] * rinv;
    __syncthreads();
    for (int j = t; j < DIM; j += 256) {
        float acc = fb1[j];
        for (int l = 0; l < LVL; ++l) {
            int slot = maskinfo[1 + l];
            if (slot >= 0) {
                const float* wrow = fw1 + (size_t)(slot << 10) * DIM;
                const float* pl = sp[l];
                for (int d = 0; d < DIM; ++d)
                    acc = fmaf(pl[d], wrow[(size_t)d * DIM + j], acc);
                acc = fmaf(actsW[l], fw1[(size_t)((k << 10) + slot) * DIM + j], acc);
            }
        }
        sh[j] = fmaxf(acc, 0.f);
    }
    __syncthreads();
    for (int j = t; j < DIM; j += 256) {
        float acc = fb2[j];
        for (int d = 0; d < DIM; ++d)
            acc = fmaf(sh[d], fw2[(size_t)d * DIM + j], acc);
        outProto[(size_t)c * DIM + j] = acc;
    }
}

// ---------------------------------------------------------------------------
// Workspace layout (bytes)
// ---------------------------------------------------------------------------
#define WS_W1P   ((size_t)0)                         // 8 MB packed bf16
#define WS_W2P   ((size_t)8388608)                   // 8 MB
#define WS_C1    ((size_t)16777216)                  // 16 KB
#define WS_PSUM  ((size_t)16793600)                  // 1 MB  (zeroed)
#define WS_RAW   ((size_t)17842176)                  // 256 KB (zeroed)
#define WS_CNT   ((size_t)18104320)                  // 256 B (zeroed, float counts)
#define WS_TC    ((size_t)18104576)                  // 4 KB  (zeroed)
#define WS_ICNT  ((size_t)18108672)                  // 256 B (zeroed, int hist)
#define WS_CUR   ((size_t)18108928)                  // 256 B (zeroed, scatter cursor)
#define WS_ACTS  ((size_t)18109184)                  // 256 B
#define WS_MASK  ((size_t)18109440)                  // 256 B
#define WS_BOFF  ((size_t)18109696)                  // 512 B (binOff[65])
#define WS_PERM  ((size_t)18110208)                  // 64 KB
#define WS_LABS  ((size_t)18175744)                  // 64 KB
#define WS_ZERO_OFF   WS_PSUM
#define WS_ZERO_BYTES (WS_ACTS - WS_PSUM)

extern "C" void kernel_launch(void* const* d_in, const int* in_sizes, int n_in,
                              void* d_out, int out_size, void* d_ws, size_t ws_size,
                              hipStream_t stream) {
    const float* X      = (const float*)d_in[0];
    const int*   lab    = (const int*)d_in[1];
    const float* sp_w1  = (const float*)d_in[2];
    const float* sp_b1  = (const float*)d_in[3];
    const float* sp_w2  = (const float*)d_in[4];
    const float* sp_b2  = (const float*)d_in[5];
    const float* lvl    = (const float*)d_in[6];
    const float* an_w1  = (const float*)d_in[7];
    const float* an_b1  = (const float*)d_in[8];
    const float* an_g   = (const float*)d_in[9];
    const float* an_bt  = (const float*)d_in[10];
    const float* an_w2  = (const float*)d_in[11];
    const float* an_b2  = (const float*)d_in[12];
    const float* fu_w1  = (const float*)d_in[13];
    const float* fu_b1  = (const float*)d_in[14];
    const float* fu_w2  = (const float*)d_in[15];
    const float* fu_b2  = (const float*)d_in[16];

    char* ws = (char*)d_ws;
    unsigned* W1p    = (unsigned*)(ws + WS_W1P);
    unsigned* W2p    = (unsigned*)(ws + WS_W2P);
    float*    c1     = (float*)(ws + WS_C1);
    float*    psum   = (float*)(ws + WS_PSUM);
    float*    raw    = (float*)(ws + WS_RAW);
    float*    cnts   = (float*)(ws + WS_CNT);
    float*    tc     = (float*)(ws + WS_TC);
    int*      icnt   = (int*)(ws + WS_ICNT);
    int*      cursor = (int*)(ws + WS_CUR);
    float*    actsW  = (float*)(ws + WS_ACTS);
    int*      minfo  = (int*)(ws + WS_MASK);
    int*      binOff = (int*)(ws + WS_BOFF);
    int*      perm   = (int*)(ws + WS_PERM);
    int*      labS   = (int*)(ws + WS_LABS);
    float*    out    = (float*)d_out;

    hipMemsetAsync(ws + WS_ZERO_OFF, 0, WS_ZERO_BYTES, stream);

    // Weight prepack (bf16 fragment order) + fused level bias
    prepack_kernel<<<8192, 256, 0, stream>>>(an_w1, W1p, 1025 * DIM);
    prepack_kernel<<<8192, 256, 0, stream>>>(an_w2, W2p, 1024 * DIM);
    c1_kernel<<<16, 256, 0, stream>>>(an_w1, an_b1, lvl, c1);

    // Counting sort of rows by label
    hist_kernel<<<N_SUP / 256, 256, 0, stream>>>(lab, icnt);
    scan_kernel<<<1, 64, 0, stream>>>(icnt, binOff, cnts);
    scatter_kernel<<<N_SUP / 256, 256, 0, stream>>>(lab, binOff, cursor, perm, labS);

    // Selector path
    colsum_kernel<<<dim3(4, 16), 256, 0, stream>>>(X, tc);
    splevel_kernel<<<1, 512, 0, stream>>>(tc, sp_w1, sp_b1, sp_w2, sp_b2,
                                          actsW, minfo, out + NCLS * DIM);

    // Raw segment sum over sorted rows
    rawsum_kernel<<<N_SUP / BM, 256, 0, stream>>>(X, perm, labS, raw);

    // Main fused adapter (WMMA) + sorted-run segment reduction
    anet_kernel<<<dim3(N_SUP / BM, LVL), 256, SMEM_BYTES, stream>>>(
        X, perm, labS, W1p, W2p, c1, an_g, an_bt, an_b2, psum);

    // Fusion MLP + output select
    fusion_kernel<<<NCLS, 256, 0, stream>>>(psum, raw, cnts, actsW, minfo,
                                            fu_w1, fu_b1, fu_w2, fu_b2, out);
}
